// FirstAttn_52252572123713
// MI455X (gfx1250) — matrix operand order repaired and checked by reading, not compiled, not run on hardware
//
#include <hip/hip_runtime.h>
#include <hip/hip_bf16.h>
#include <math.h>

// ---------------------------------------------------------------------------
// Problem constants: B=2, C_in=32, C_id=256, H=W=256, 8x8 windows -> 2048
// windows, 8 heads, head dim 32, 64 tokens/window.
// ---------------------------------------------------------------------------
#define HW      65536
#define CIN     32
#define CID     256
#define NB      2
#define NWB     2048
#define NHEAD   8
#define HD      32
#define NTOK    64
#define ATTN_ELEMS 67108864ull  // 2048*8*64*64

typedef __attribute__((ext_vector_type(16))) __bf16 v16bf;
typedef __attribute__((ext_vector_type(8)))  __bf16 v8bf;
typedef __attribute__((ext_vector_type(8)))  float  v8f;
typedef __attribute__((ext_vector_type(4)))  int    v4i;

#define ASG __attribute__((address_space(1)))
#define ASL __attribute__((address_space(3)))

#if __has_builtin(__builtin_amdgcn_global_load_async_to_lds_b128)
#define HAVE_ASYNC_LDS 1
#else
#define HAVE_ASYNC_LDS 0
#endif

__device__ __forceinline__ void wait_asynccnt0() {
#if __has_builtin(__builtin_amdgcn_s_wait_asynccnt)
    __builtin_amdgcn_s_wait_asynccnt(0);
#else
    asm volatile("s_wait_asynccnt 0" ::: "memory");
#endif
}

// CDNA5 wave32 WMMA: D(f32 16x16) = A(bf16 16x32) * B(bf16 32x16) + C
__device__ __forceinline__ v8f wmma_bf16(v16bf a, v16bf b, v8f c) {
    return __builtin_amdgcn_wmma_f32_16x16x32_bf16(
        false, a, false, b, (short)0, c, false, false);
}

// A-matrix 16x32 (ISA 7.12.2): lane holds row M=lane%16; its 16 elements are
// two contiguous K-runs [half*8, +8) and [16+half*8, +8). Two b128 loads.
__device__ __forceinline__ v16bf frag_rowA(const __bf16* __restrict__ row, int half) {
    const v8bf* r = (const v8bf*)row;
    v8bf lo = r[half];
    v8bf hi = r[half + 2];
    return __builtin_shufflevector(lo, hi, 0,1,2,3,4,5,6,7,8,9,10,11,12,13,14,15);
}
// B-matrix 32x16: lane holds col N=lane%16; its 16 elements are two
// contiguous K-runs [half*8, +8) and [16+half*8, +8). Two b128 loads.
__device__ __forceinline__ v16bf frag_rowB(const __bf16* __restrict__ row, int half) {
    const v8bf* r = (const v8bf*)row;
    v8bf lo = r[half];
    v8bf hi = r[half + 2];
    return __builtin_shufflevector(lo, hi, 0,1,2,3,4,5,6,7,8,9,10,11,12,13,14,15);
}

// ---------------------------------------------------------------------------
// f32 -> bf16 row-major weight conversion (tiny: 16KB / 128KB)
// ---------------------------------------------------------------------------
__global__ void f32_to_bf16_kernel(const float* __restrict__ src,
                                   __bf16* __restrict__ dst, int n) {
    int i = blockIdx.x * blockDim.x + threadIdx.x;
    if (i < n) dst[i] = (__bf16)src[i];
}

// ---------------------------------------------------------------------------
// Kernel 1: channel LayerNorm2d (C=32, biased var, eps=1e-5).
// Reads NCHW fp32, writes PIXEL-MAJOR bf16 X[p][32] so the conv1x1
// B-fragment becomes two contiguous global_load_b128 per lane.
// ---------------------------------------------------------------------------
__global__ void ln2d_kernel(const float* __restrict__ x, const float* __restrict__ g,
                            __bf16* __restrict__ out) {
    int p = blockIdx.x * blockDim.x + threadIdx.x;
    if (p >= NB * HW) return;
    int b = p >> 16, hw = p & (HW - 1);
    size_t base = (size_t)b * CIN * HW + hw;
    float vals[CIN];
    float s = 0.f, s2 = 0.f;
    #pragma unroll
    for (int c = 0; c < CIN; ++c) {
        float v = x[base + (size_t)c * HW];
        vals[c] = v; s += v; s2 += v * v;
    }
    float mean = s * (1.f / CIN);
    float var  = s2 * (1.f / CIN) - mean * mean;
    float rstd = rsqrtf(var + 1e-5f);
    #pragma unroll
    for (int c = 0; c < CIN; ++c)
        out[(size_t)p * CIN + c] = (__bf16)((vals[c] - mean) * rstd * g[c]);
}

// ---------------------------------------------------------------------------
// Kernel 2: 1x1 conv as GEMM, one 16x16x32 WMMA per tile (K=32 exactly).
// A = bf16 weights row-major [256][32], B = pixel-major X rows. fp32 out.
// ---------------------------------------------------------------------------
__global__ __launch_bounds__(256) void conv1x1_wmma_kernel(
        const __bf16* __restrict__ Xpm, const __bf16* __restrict__ Wb,
        const float* __restrict__ bias, float* __restrict__ out) {
    int wid  = threadIdx.x >> 5, lane = threadIdx.x & 31;
    int half = lane >> 4, lm = lane & 15;
    int id = blockIdx.x * 8 + wid;          // NB * 16 * 4096 tiles
    int bb = id >> 16;
    int r  = id & 65535;
    int mt = r & 15;
    int p0 = (r >> 4) << 4;
    v16bf a  = frag_rowA(&Wb[(mt * 16 + lm) * CIN], half);
    v16bf bm = frag_rowB(&Xpm[(size_t)(bb * HW + p0 + lm) * CIN], half);
    v8f c = {};
    c = wmma_bf16(a, bm, c);
    #pragma unroll
    for (int rr = 0; rr < 8; ++rr) {        // C/D layout: M = rr + 8*half, N = lm
        int m = mt * 16 + rr + 8 * half;
        out[((size_t)(bb * CID + m)) * HW + p0 + lm] = c[rr] + bias[m];
    }
}

// ---------------------------------------------------------------------------
// Kernel 3: depthwise 3x3 conv, pad=1, fp32 in -> bf16 out.
// ---------------------------------------------------------------------------
__global__ void dwconv3_kernel(const float* __restrict__ x, const float* __restrict__ wt,
                               const float* __restrict__ bias, __bf16* __restrict__ out) {
    size_t idx = (size_t)blockIdx.x * blockDim.x + threadIdx.x;
    if (idx >= (size_t)NB * CID * HW) return;
    int hw = (int)(idx & (HW - 1));
    int c  = (int)((idx >> 16) & (CID - 1));
    int h = hw >> 8, w = hw & 255;
    size_t base = idx - hw;
    float acc = bias[c];
    #pragma unroll
    for (int dy = -1; dy <= 1; ++dy) {
        int hh = h + dy;
        if (hh < 0 || hh > 255) continue;
        #pragma unroll
        for (int dx = -1; dx <= 1; ++dx) {
            int ww = w + dx;
            if (ww < 0 || ww > 255) continue;
            acc += x[base + hh * 256 + ww] * wt[c * 9 + (dy + 1) * 3 + (dx + 1)];
        }
    }
    out[idx] = (__bf16)acc;
}

// ---------------------------------------------------------------------------
// Kernel 4: windowed attention, one block (128 thr = 4 waves) per (window,head).
// q/k row-major [tok][dim] (QK^T fragments = contiguous LDS b128 runs);
// v TRANSPOSED [dim][tok] so PV B-fragments are contiguous too.  V is staged
// with GLOBAL_LOAD_ASYNC_TO_LDS_B128 (16B global chunk = one vsT row
// segment), drained by s_wait_asynccnt.
// ---------------------------------------------------------------------------
__global__ __launch_bounds__(128) void window_attn_kernel(
        const __bf16* __restrict__ Q, const __bf16* __restrict__ K,
        const __bf16* __restrict__ V, float* __restrict__ attn_out,
        __bf16* __restrict__ Y) {
    __shared__ alignas(16) __bf16 qs[NTOK][HD];
    __shared__ alignas(16) __bf16 ks[NTOK][HD];
    __shared__ alignas(16) __bf16 vsT[HD][NTOK];
    __shared__ alignas(16) float  S[NTOK][NTOK];
    __shared__ alignas(16) __bf16 Pb[NTOK][NTOK];

    int wb = blockIdx.x >> 3, hd = blockIdx.x & 7;
    int b = wb >> 10, hb = (wb >> 5) & 31, wx = wb & 31;
    int wid = threadIdx.x >> 5, lane = threadIdx.x & 31;
    int half = lane >> 4, lm = lane & 15;

    // ---- stage V: 256 16-byte chunks (dim d, window-row wr) -> vsT[d][wr*8..]
#if HAVE_ASYNC_LDS
    for (int ch = threadIdx.x; ch < 256; ch += 128) {
        int d = ch >> 3, wr = ch & 7;
        const __bf16* gp = &V[((size_t)(b * CID + hd * HD + d)) * HW
                              + (hb * 8 + wr) * 256 + wx * 8];
        __builtin_amdgcn_global_load_async_to_lds_b128(
            (ASG v4i*)(void*)gp, (ASL v4i*)(void*)&vsT[d][wr * 8], 0, 0);
    }
    wait_asynccnt0();
#else
    for (int ch = threadIdx.x; ch < 256; ch += 128) {
        int d = ch >> 3, wr = ch & 7;
        const v8bf* gp = (const v8bf*)&V[((size_t)(b * CID + hd * HD + d)) * HW
                                         + (hb * 8 + wr) * 256 + wx * 8];
        v8bf v = *gp;
        #pragma unroll
        for (int j = 0; j < 8; ++j) vsT[d][wr * 8 + j] = v[j];
    }
#endif

    // ---- stage Q, K row-major: vector 16B global loads, scalar LDS scatter
    for (int ch = threadIdx.x; ch < 512; ch += 128) {
        int tens = ch >> 8;                     // 0 = Q, 1 = K
        int d = (ch >> 3) & 31, wr = ch & 7;
        const __bf16* src = tens ? K : Q;
        __bf16 (*dst)[HD] = tens ? ks : qs;
        const v8bf* gp = (const v8bf*)&src[((size_t)(b * CID + hd * HD + d)) * HW
                                           + (hb * 8 + wr) * 256 + wx * 8];
        v8bf v = *gp;
        #pragma unroll
        for (int j = 0; j < 8; ++j) dst[wr * 8 + j][d] = v[j];
    }
    __syncthreads();

    // ---- S(64x64) = q * k^T : 16 tiles over 4 waves (wave-uniform trip)
    for (int t = wid; t < 16; t += 4) {
        int ti = t >> 2, tj = t & 3;
        v16bf a  = frag_rowA(qs[ti * 16 + lm], half);
        v16bf bm = frag_rowB(ks[tj * 16 + lm], half);   // B[d][m] = k[m][d]
        v8f c = {};
        c = wmma_bf16(a, bm, c);
        #pragma unroll
        for (int rr = 0; rr < 8; ++rr)
            S[ti * 16 + rr + 8 * half][tj * 16 + lm] = c[rr];
    }
    __syncthreads();

    // ---- fp32 softmax per row; emit attn (output #1) + bf16 P
    if (threadIdx.x < NTOK) {
        int row = threadIdx.x;
        float mx = -__builtin_inff();
        for (int m = 0; m < NTOK; ++m) mx = fmaxf(mx, S[row][m]);
        float sum = 0.f;
        for (int m = 0; m < NTOK; ++m) {
            float ex = __expf(S[row][m] - mx);
            S[row][m] = ex; sum += ex;
        }
        float inv = 1.f / sum;
        size_t ab = (((size_t)wb * NHEAD + hd) * NTOK + row) * NTOK;
        for (int m = 0; m < NTOK; ++m) {
            float pv = S[row][m] * inv;
            attn_out[ab + m] = pv;
            Pb[row][m] = (__bf16)pv;
        }
    }
    __syncthreads();

    // ---- O(64x32) = P(64x64) * v : 8 tiles, K=64 -> 2 WMMA steps
    for (int t = wid; t < 8; t += 4) {
        int ti = t >> 1, tj = t & 1;
        v8f c = {};
        #pragma unroll
        for (int kk = 0; kk < 2; ++kk) {
            v16bf a  = frag_rowA(&Pb[ti * 16 + lm][kk * 32], half);
            v16bf bm = frag_rowB(&vsT[tj * 16 + lm][kk * 32], half); // B[k][n]=vsT[n][k]
            c = wmma_bf16(a, bm, c);
        }
        #pragma unroll
        for (int rr = 0; rr < 8; ++rr) {
            int n = ti * 16 + rr + 8 * half;
            int d = tj * 16 + lm;
            Y[((size_t)wb * NTOK + n) * CID + hd * HD + d] = (__bf16)c[rr];
        }
    }
}

// ---------------------------------------------------------------------------
// Kernel 5: projection GEMM (K=256, 8 WMMAs) + faithful window_reverse
// scramble: out[b, (co%4)*64+n, hb*8+(co/4)/8, wx*8+(co/4)%8] = y2[wb*64+n, co]
// ---------------------------------------------------------------------------
__global__ __launch_bounds__(128) void proj_scatter_kernel(
        const __bf16* __restrict__ Yin, const __bf16* __restrict__ Wpb,
        const float* __restrict__ bp, float* __restrict__ out) {
    int wid = threadIdx.x >> 5, lane = threadIdx.x & 31;
    int half = lane >> 4, lm = lane & 15;
    int id = blockIdx.x * 4 + wid;          // 8192 M-tiles * 16 N-tiles
    int mt = id >> 4, nt = id & 15;
    int p0 = mt * 16, n0 = nt * 16;
    v8f c = {};
    #pragma unroll
    for (int kc = 0; kc < 8; ++kc) {
        v16bf a  = frag_rowA(&Yin[(size_t)(p0 + lm) * CID + kc * 32], half);
        v16bf bm = frag_rowB(&Wpb[(size_t)(n0 + lm) * CID + kc * 32], half);
        c = wmma_bf16(a, bm, c);
    }
    #pragma unroll
    for (int rr = 0; rr < 8; ++rr) {
        int p  = p0 + rr + 8 * half;        // flattened (wb, n)
        int co = n0 + lm;
        float v = c[rr] + bp[co];
        int wb = p >> 6, n = p & 63;
        int c2 = (co & 3) * 64 + n;
        int cq = co >> 2;
        int i = cq >> 3, j = cq & 7;
        int bb = wb >> 10, hbw = (wb >> 5) & 31, wxw = wb & 31;
        int h = hbw * 8 + i, w = wxw * 8 + j;
        out[((size_t)(bb * CID + c2)) * HW + h * 256 + w] = v;
    }
}

// ---------------------------------------------------------------------------
// Orchestration. Workspace layout (bytes):
//   0         : lnX_lms  bf16 pixel-major   8,388,608
//   8388608   : lnX_pan  bf16 pixel-major   8,388,608
//   16777216  : t        f32              134,217,728  (conv1x1 out, reused)
//   150994944 : qdw      bf16              67,108,864
//   218103808 : kdw      bf16              67,108,864
//   285212672 : vdw      bf16              67,108,864
//   352321536 : y        bf16              67,108,864
//   419430400 : wconv_b  bf16                  16,384  (reused q/k/v)
//   419446784 : wproj_b  bf16                 131,072   (total ~419.6 MB)
// d_out = [attn fp32 (2048,8,64,64)] ++ [out fp32 (2,256,256,256)]
// ---------------------------------------------------------------------------
extern "C" void kernel_launch(void* const* d_in, const int* in_sizes, int n_in,
                              void* d_out, int out_size, void* d_ws, size_t ws_size,
                              hipStream_t stream) {
    (void)in_sizes; (void)n_in; (void)out_size; (void)ws_size;
    const float* lms    = (const float*)d_in[0];
    const float* pan    = (const float*)d_in[1];
    const float* g_ms   = (const float*)d_in[2];
    const float* g_pan  = (const float*)d_in[3];
    const float* wq1    = (const float*)d_in[4];
    const float* bq1    = (const float*)d_in[5];
    const float* wq2    = (const float*)d_in[6];
    const float* bq2    = (const float*)d_in[7];
    const float* wk1    = (const float*)d_in[8];
    const float* bk1    = (const float*)d_in[9];
    const float* wk2    = (const float*)d_in[10];
    const float* bk2    = (const float*)d_in[11];
    const float* wv1    = (const float*)d_in[12];
    const float* bv1    = (const float*)d_in[13];
    const float* wv2    = (const float*)d_in[14];
    const float* bv2    = (const float*)d_in[15];
    const float* w_proj = (const float*)d_in[16];
    const float* b_proj = (const float*)d_in[17];

    char* ws = (char*)d_ws;
    __bf16* ln_lms  = (__bf16*)(ws + 0ull);
    __bf16* ln_pan  = (__bf16*)(ws + 8388608ull);
    float*  t       = (float*) (ws + 16777216ull);
    __bf16* qdw     = (__bf16*)(ws + 150994944ull);
    __bf16* kdw     = (__bf16*)(ws + 218103808ull);
    __bf16* vdw     = (__bf16*)(ws + 285212672ull);
    __bf16* yb      = (__bf16*)(ws + 352321536ull);
    __bf16* wconv_b = (__bf16*)(ws + 419430400ull);
    __bf16* wproj_b = (__bf16*)(ws + 419446784ull);

    float* attn = (float*)d_out;
    float* outp = (float*)d_out + ATTN_ELEMS;

    ln2d_kernel<<<512, 256, 0, stream>>>(lms, g_ms, ln_lms);
    ln2d_kernel<<<512, 256, 0, stream>>>(pan, g_pan, ln_pan);

    // Q path
    f32_to_bf16_kernel<<<32, 256, 0, stream>>>(wq1, wconv_b, CID * CIN);
    conv1x1_wmma_kernel<<<16384, 256, 0, stream>>>(ln_pan, wconv_b, bq1, t);
    dwconv3_kernel<<<131072, 256, 0, stream>>>(t, wq2, bq2, qdw);
    // K path
    f32_to_bf16_kernel<<<32, 256, 0, stream>>>(wk1, wconv_b, CID * CIN);
    conv1x1_wmma_kernel<<<16384, 256, 0, stream>>>(ln_lms, wconv_b, bk1, t);
    dwconv3_kernel<<<131072, 256, 0, stream>>>(t, wk2, bk2, kdw);
    // V path
    f32_to_bf16_kernel<<<32, 256, 0, stream>>>(wv1, wconv_b, CID * CIN);
    conv1x1_wmma_kernel<<<16384, 256, 0, stream>>>(ln_lms, wconv_b, bv1, t);
    dwconv3_kernel<<<131072, 256, 0, stream>>>(t, wv2, bv2, vdw);

    window_attn_kernel<<<16384, 128, 0, stream>>>(qdw, kdw, vdw, attn, yb);

    f32_to_bf16_kernel<<<256, 256, 0, stream>>>(w_proj, wproj_b, CID * CID);
    proj_scatter_kernel<<<32768, 128, 0, stream>>>(yb, wproj_b, b_proj, outp);
}
